// SelfAttention_57853209477191
// MI455X (gfx1250) — matrix operand order repaired and checked by reading, not compiled, not run on hardware
//
#include <hip/hip_runtime.h>

// Problem constants (match reference)
#define ED   1024   // EMBED_DIM
#define HD   1024   // HIDDEN_DIM
#define BB   4
#define NN   2048

typedef __attribute__((ext_vector_type(16))) __bf16 v16bf;
typedef __attribute__((ext_vector_type(8)))  __bf16 v8bf;
typedef __attribute__((ext_vector_type(8)))  float  v8f;

static __device__ __forceinline__ v16bf make16(v8bf lo, v8bf hi) {
    return __builtin_shufflevector(lo, hi, 0,1,2,3,4,5,6,7,8,9,10,11,12,13,14,15);
}

static __device__ __forceinline__ v8bf cvt8(float4 a, float4 b) {
    v8bf r;
    r[0] = (__bf16)a.x; r[1] = (__bf16)a.y; r[2] = (__bf16)a.z; r[3] = (__bf16)a.w;
    r[4] = (__bf16)b.x; r[5] = (__bf16)b.y; r[6] = (__bf16)b.z; r[7] = (__bf16)b.w;
    return r;
}

static __device__ __forceinline__ v8f wmma_bf16(v16bf a, v16bf b, v8f c) {
    return __builtin_amdgcn_wmma_f32_16x16x32_bf16(
        /*neg_a=*/false, a, /*neg_b=*/false, b,
        /*c_mod=*/(short)0, c, /*reuse_a=*/false, /*reuse_b=*/false);
}

// ---------------------------------------------------------------------------
// 0) x f32 -> bf16, one pass (removes redundant per-wave conversion work)
// ---------------------------------------------------------------------------
__global__ __launch_bounds__(256)
void cvt_x(const float* __restrict__ x, __bf16* __restrict__ xb) {
    size_t i = ((size_t)blockIdx.x * 256 + threadIdx.x) * 8;
    const float4* p = (const float4*)(x + i);
    *(v8bf*)(xb + i) = cvt8(p[0], p[1]);
}

// ---------------------------------------------------------------------------
// 1) W [E][H] f32  ->  Wt [H][E] bf16   (transpose + convert)
// ---------------------------------------------------------------------------
__global__ __launch_bounds__(256)
void transpose_cvt(const float* __restrict__ W, __bf16* __restrict__ Wt) {
    int idx = blockIdx.x * 256 + threadIdx.x;     // 0 .. E*H-1
    int e = idx >> 10;                            // / HD
    int h = idx & (HD - 1);
    Wt[(size_t)h * ED + e] = (__bf16)W[idx];
}

// ---------------------------------------------------------------------------
// 2) QKV projection: out = xb @ Wt^T + b   (all-bf16 operands, fp32 accum)
//    Each wave: one 16-row tile x FOUR 16-col tiles (A reused 4x).
//    mode 0: row-major bf16 [B*N][H] (Q, K);  mode 1: transposed [B][H][N] (V^T)
// ---------------------------------------------------------------------------
__global__ __launch_bounds__(128)
void qkv_gemm(const __bf16* __restrict__ xb,    // [B*N][E] bf16
              const __bf16* __restrict__ Wt,    // [H][E]  bf16
              const float* __restrict__ bias,   // [H]
              __bf16* __restrict__ outRM,       // mode 0
              __bf16* __restrict__ outTR,       // mode 1
              int mode) {
    int lane = threadIdx.x & 31;
    int wv   = threadIdx.x >> 5;
    int n    = lane & 15;           // A m-index == C/D n-index
    int half = lane >> 4;
    int row0    = blockIdx.x * 16;
    int colBase = (blockIdx.y * 16 + wv * 4) * 16;   // 4 tiles of 16 cols

    const __bf16* arow = xb + (size_t)(row0 + n) * ED;
    const __bf16* brow[4];
    #pragma unroll
    for (int j = 0; j < 4; ++j)
        brow[j] = Wt + (size_t)(colBase + j * 16 + n) * ED;

    v8f acc[4];
    #pragma unroll
    for (int j = 0; j < 4; ++j) acc[j] = (v8f){0.f,0.f,0.f,0.f,0.f,0.f,0.f,0.f};

    for (int e0 = 0; e0 < ED; e0 += 32) {
        __builtin_prefetch(arow + e0 + 256, 0, 0);
        v16bf a = make16(*(const v8bf*)(arow + e0 + half * 8),
                         *(const v8bf*)(arow + e0 + 16 + half * 8));
        #pragma unroll
        for (int j = 0; j < 4; ++j) {
            const v8bf* pb = (const v8bf*)(brow[j] + e0 + 8 * half);
            acc[j] = wmma_bf16(a, make16(pb[0], pb[2]), acc[j]);
        }
    }

    if (mode == 0) {
        #pragma unroll
        for (int j = 0; j < 4; ++j) {
            float bv = bias[colBase + j * 16 + n];
            #pragma unroll
            for (int v = 0; v < 8; ++v) {
                int m = v + 8 * half;   // D: lane n, VGPR v -> m = v + 8*half
                outRM[(size_t)(row0 + m) * HD + colBase + j * 16 + n] =
                    (__bf16)(acc[j][v] + bv);
            }
        }
    } else {
        int b    = row0 / NN;
        int key0 = (row0 % NN) + 8 * half;
        #pragma unroll
        for (int j = 0; j < 4; ++j) {
            float bv = bias[colBase + j * 16 + n];
            v8bf pack;
            #pragma unroll
            for (int v = 0; v < 8; ++v) pack[v] = (__bf16)(acc[j][v] + bv);
            *(v8bf*)(outTR + ((size_t)b * HD + colBase + j * 16 + n) * NN + key0) = pack;
        }
    }
}

// ---------------------------------------------------------------------------
// 3) scores S[q][k] = (Q[q,:] . K[k,:]) * rscale   (one batch per launch)
//    Each wave: one q tile x FOUR k tiles.
// ---------------------------------------------------------------------------
__global__ __launch_bounds__(128)
void scores_kernel(const __bf16* __restrict__ Q,  // [N][H] (batch slice)
                   const __bf16* __restrict__ K,  // [N][H]
                   float* __restrict__ S,         // [N][N]
                   float rscale) {
    int lane = threadIdx.x & 31;
    int wv   = threadIdx.x >> 5;
    int n    = lane & 15;
    int half = lane >> 4;
    int q0    = blockIdx.x * 16;
    int kBase = (blockIdx.y * 16 + wv * 4) * 16;

    const __bf16* qrow = Q + (size_t)(q0 + n) * HD;
    const __bf16* krow[4];
    #pragma unroll
    for (int j = 0; j < 4; ++j)
        krow[j] = K + (size_t)(kBase + j * 16 + n) * HD;

    v8f acc[4];
    #pragma unroll
    for (int j = 0; j < 4; ++j) acc[j] = (v8f){0.f,0.f,0.f,0.f,0.f,0.f,0.f,0.f};

    for (int h0 = 0; h0 < HD; h0 += 32) {
        __builtin_prefetch(qrow + h0 + 256, 0, 0);
        v16bf a = make16(*(const v8bf*)(qrow + h0 + half * 8),
                         *(const v8bf*)(qrow + h0 + 16 + half * 8));
        #pragma unroll
        for (int j = 0; j < 4; ++j) {
            const v8bf* pb = (const v8bf*)(krow[j] + h0 + 8 * half);
            acc[j] = wmma_bf16(a, make16(pb[0], pb[2]), acc[j]);
        }
    }

    #pragma unroll
    for (int j = 0; j < 4; ++j) {
        #pragma unroll
        for (int v = 0; v < 8; ++v) {
            int m = v + 8 * half;
            S[(size_t)(q0 + m) * NN + kBase + j * 16 + n] = acc[j][v] * rscale;
        }
    }
}

// ---------------------------------------------------------------------------
// 4) Row softmax over S[row][0..N-1]; write bf16 P in place over the row's
//    fp32 bytes (first half of the row) — aliasing confined within the row.
// ---------------------------------------------------------------------------
__global__ __launch_bounds__(256)
void softmax_kernel(float* __restrict__ S) {
    __shared__ float red[256];
    int t = threadIdx.x;
    float* srow = S + (size_t)blockIdx.x * NN;

    float r[8];
    #pragma unroll
    for (int i = 0; i < 8; ++i) r[i] = srow[t + 256 * i];

    float m = r[0];
    #pragma unroll
    for (int i = 1; i < 8; ++i) m = fmaxf(m, r[i]);
    red[t] = m;
    __syncthreads();
    for (int s = 128; s > 0; s >>= 1) {
        if (t < s) red[t] = fmaxf(red[t], red[t + s]);
        __syncthreads();
    }
    float M = red[0];
    __syncthreads();

    float e[8]; float sum = 0.f;
    #pragma unroll
    for (int i = 0; i < 8; ++i) { e[i] = __expf(r[i] - M); sum += e[i]; }
    red[t] = sum;
    __syncthreads();
    for (int s = 128; s > 0; s >>= 1) {
        if (t < s) red[t] += red[t + s];
        __syncthreads();
    }
    float inv = 1.0f / red[0];

    __bf16* prow = (__bf16*)srow;
    #pragma unroll
    for (int i = 0; i < 8; ++i) prow[t + 256 * i] = (__bf16)(e[i] * inv);
}

// ---------------------------------------------------------------------------
// 5) O = P @ V : P bf16 rows embedded in S (stride NN floats), V^T [H][N].
//    Each wave: one q tile x FOUR output-column tiles.
// ---------------------------------------------------------------------------
__global__ __launch_bounds__(128)
void pv_kernel(const float* __restrict__ S,      // rows hold bf16 P
               const __bf16* __restrict__ Vt,    // [H][N] (batch slice)
               float* __restrict__ out) {        // [N][H] (batch slice)
    int lane = threadIdx.x & 31;
    int wv   = threadIdx.x >> 5;
    int n    = lane & 15;
    int half = lane >> 4;
    int q0    = blockIdx.x * 16;
    int cBase = (blockIdx.y * 16 + wv * 4) * 16;

    const __bf16* prow = (const __bf16*)(S + (size_t)(q0 + n) * NN);
    const __bf16* vrow[4];
    #pragma unroll
    for (int j = 0; j < 4; ++j)
        vrow[j] = Vt + (size_t)(cBase + j * 16 + n) * NN;

    v8f acc[4];
    #pragma unroll
    for (int j = 0; j < 4; ++j) acc[j] = (v8f){0.f,0.f,0.f,0.f,0.f,0.f,0.f,0.f};

    for (int k0 = 0; k0 < NN; k0 += 32) {
        __builtin_prefetch(prow + k0 + 256, 0, 0);
        v16bf a = make16(*(const v8bf*)(prow + k0 + half * 8),
                         *(const v8bf*)(prow + k0 + 16 + half * 8));
        #pragma unroll
        for (int j = 0; j < 4; ++j) {
            const v8bf* pb = (const v8bf*)(vrow[j] + k0 + 8 * half);
            acc[j] = wmma_bf16(a, make16(pb[0], pb[2]), acc[j]);
        }
    }

    #pragma unroll
    for (int j = 0; j < 4; ++j) {
        #pragma unroll
        for (int v = 0; v < 8; ++v) {
            int m = v + 8 * half;
            out[(size_t)(q0 + m) * HD + cBase + j * 16 + n] = acc[j][v];
        }
    }
}

// ---------------------------------------------------------------------------
extern "C" void kernel_launch(void* const* d_in, const int* in_sizes, int n_in,
                              void* d_out, int out_size, void* d_ws, size_t ws_size,
                              hipStream_t stream) {
    const float* x  = (const float*)d_in[0];
    const float* Wq = (const float*)d_in[1];
    const float* bq = (const float*)d_in[2];
    const float* Wk = (const float*)d_in[3];
    const float* bk = (const float*)d_in[4];
    const float* Wv = (const float*)d_in[5];
    const float* bv = (const float*)d_in[6];
    float* out = (float*)d_out;

    // Workspace carve-up (all offsets well aligned)
    char* w = (char*)d_ws;
    size_t off = 0;
    __bf16* xb = (__bf16*)(w + off); off += (size_t)BB * NN * ED * 2;  // 16 MB
    __bf16* Wt = (__bf16*)(w + off); off += (size_t)3 * ED * HD * 2;   //  6 MB
    __bf16* Qb = (__bf16*)(w + off); off += (size_t)BB * NN * HD * 2;  // 16 MB
    __bf16* Kb = (__bf16*)(w + off); off += (size_t)BB * NN * HD * 2;  // 16 MB
    __bf16* Vt = (__bf16*)(w + off); off += (size_t)BB * HD * NN * 2;  // 16 MB
    float*  S  = (float*) (w + off);                                   // 16 MB (per-batch reuse)

    // 0) x -> bf16 once
    cvt_x<<<(BB * NN * ED) / (256 * 8), 256, 0, stream>>>(x, xb);

    // 1) weight transpose+convert
    int wblocks = (ED * HD) / 256;
    transpose_cvt<<<wblocks, 256, 0, stream>>>(Wq, Wt + (size_t)0 * ED * HD);
    transpose_cvt<<<wblocks, 256, 0, stream>>>(Wk, Wt + (size_t)1 * ED * HD);
    transpose_cvt<<<wblocks, 256, 0, stream>>>(Wv, Wt + (size_t)2 * ED * HD);

    // 2) Q, K (row-major), V (transposed). 256 output cols per block.
    dim3 ggrid((BB * NN) / 16, HD / 256);
    qkv_gemm<<<ggrid, 128, 0, stream>>>(xb, Wt + (size_t)0 * ED * HD, bq, Qb, nullptr, 0);
    qkv_gemm<<<ggrid, 128, 0, stream>>>(xb, Wt + (size_t)1 * ED * HD, bk, Kb, nullptr, 0);
    qkv_gemm<<<ggrid, 128, 0, stream>>>(xb, Wt + (size_t)2 * ED * HD, bv, nullptr, Vt, 1);

    const float rscale = 1.0f / (32.0f + 1e-8f);  // 1/(sqrt(1024)+eps)

    // 3-5) per-batch attention (batches serialized on the stream; S reused)
    for (int b = 0; b < BB; ++b) {
        scores_kernel<<<dim3(NN / 16, NN / 256), 128, 0, stream>>>(
            Qb + (size_t)b * NN * HD, Kb + (size_t)b * NN * HD, S, rscale);
        softmax_kernel<<<NN, 256, 0, stream>>>(S);
        pv_kernel<<<dim3(NN / 16, HD / 256), 128, 0, stream>>>(
            S, Vt + (size_t)b * HD * NN, out + (size_t)b * NN * HD);
    }
}
